// Encoding_32469952758287
// MI455X (gfx1250) — compile-verified, matrix operand order checked by reading, not run on hardware
//
#include <hip/hip_runtime.h>
#include <cstdint>

typedef __attribute__((ext_vector_type(16))) __bf16 v16bf;
typedef __attribute__((ext_vector_type(8)))  __bf16 v8bf;
typedef __attribute__((ext_vector_type(4)))  __bf16 v4bf;
typedef __attribute__((ext_vector_type(8)))  float  v8f;
typedef __attribute__((ext_vector_type(4)))  unsigned int v4u;
typedef __attribute__((ext_vector_type(8)))  int v8i;
typedef __attribute__((ext_vector_type(4)))  int v4i;

#define BB 32
#define NN 4096
#define DD 512
#define KK 64
#define ROWS 32            // rows of x per inner iteration
#define CHUNK 256          // rows per workgroup
#define ITERS (CHUNK / ROWS)
#define NTHREADS 256       // 8 wave32s
#define EPS 1e-12f

// ---- LDS layout (bytes, all 16B aligned) --------------------------------
#define OFF_CODES   0          // K*D bf16            = 65536
#define OFF_XF      65536      // 2 * ROWS*D f32      = 131072 (TDM dbl buf)
#define OFF_XSB     196608     // ROWS*D bf16 (row-major)   = 32768
#define OFF_XSBT    229376     // D*ROWS bf16 (d-major)     = 32768
#define OFF_WBT     262144     // K*ROWS bf16 (k-major)     = 4096
#define OFF_WT      266240     // ROWS*K f32                = 8192
#define OFF_C2      274432     // K f32
#define OFF_SC      274688     // K f32
#define OFF_X2      274944     // ROWS f32
#define OFF_SW      275072     // K f32
#define SHMEM_BYTES 275328

#if defined(__gfx1250__) && __has_builtin(__builtin_amdgcn_tensor_load_to_lds)
#define HAVE_TDM 1
#else
#define HAVE_TDM 0
#endif

// Element e of a 16-bit WMMA A/B fragment maps to K-offset:
//   ((e<8)?0:16) + 8*(lane/16) + (e&7)       (ISA 7.12.2)
// => per lane the fragment is two contiguous 8-element (16B) runs.
__device__ __forceinline__ v16bf ldfrag(const __bf16* p, int elemBase, int half) {
    const v8bf lo = *(const v8bf*)(p + elemBase + half * 8);        // ds_load_b128
    const v8bf hi = *(const v8bf*)(p + elemBase + 16 + half * 8);   // ds_load_b128
    return __builtin_shufflevector(lo, hi, 0, 1, 2, 3, 4, 5, 6, 7,
                                   8, 9, 10, 11, 12, 13, 14, 15);
}

#if HAVE_TDM
// Linear 64KB DMA: 16384 f32 from global -> LDS via the Tensor Data Mover.
// D# per ISA 8.3/8.4: group0 = {count, lds_addr, global_addr, type=2},
// group1 = {data_size=4B, tensor_dim0=tile_dim0=16384, dims1=1, stride=16384}.
// clang-23 toolchain signature: (v4u, v8i, v4i, v4i, v8i, i32 cpol).
__device__ __forceinline__ void tdm_load_tile(const float* gsrc, unsigned lds_off) {
    const unsigned long long ga = (unsigned long long)(uintptr_t)gsrc;
    v4u g0;
    g0[0] = 1u;                                            // count=1, user desc
    g0[1] = lds_off;                                       // LDS byte address
    g0[2] = (unsigned)(ga & 0xFFFFFFFFu);                  // global_addr[31:0]
    g0[3] = (unsigned)((ga >> 32) & 0x01FFFFFFu) | (2u << 30); // [56:32] | type=2
    v8i g1;
    g1[0] = 0x00020000;          // workgroup_mask=0, data_size=2 (4 bytes)
    g1[1] = (int)(16384u << 16); // tensor_dim0[15:0]=16384 in bits[63:48]
    g1[2] = (int)(1u << 16);     // tensor_dim0[31:16]=0, tensor_dim1=1
    g1[3] = (int)(16384u << 16); // tensor_dim1 hi=0, tile_dim0=16384
    g1[4] = 1;                   // tile_dim1=1, tile_dim2=0
    g1[5] = 16384;               // tensor_dim0_stride lo
    g1[6] = 0;                   // stride hi, dim1_stride lo
    g1[7] = 0;
    const v4i z4 = {0, 0, 0, 0};              // groups 2/3 unused (<=2D tensor)
    const v8i z8 = {0, 0, 0, 0, 0, 0, 0, 0};  // extra group (clang-23 form)
    __builtin_amdgcn_tensor_load_to_lds(g0, g1, z4, z4, z8, 0);
}
#endif

// --------------------------------------------------------------------------
// Kernel 0: zero the accumulators (d_out is poisoned before each timing run).
// --------------------------------------------------------------------------
extern "C" __global__ void enc_zero(float* __restrict__ Eacc,
                                    float* __restrict__ sumW,
                                    int ne, int ns) {
    const int i      = blockIdx.x * blockDim.x + threadIdx.x;
    const int stride = gridDim.x * blockDim.x;
    for (int j = i; j < ne; j += stride) Eacc[j] = 0.0f;
    for (int j = i; j < ns; j += stride) sumW[j] = 0.0f;
}

// --------------------------------------------------------------------------
// Kernel 1: fused  dot -> dist -> softmax -> W^T x  in a single pass over x.
// One workgroup = one (batch, 256-row chunk). 8 waves, TDM double buffering.
// --------------------------------------------------------------------------
extern "C" __global__ __launch_bounds__(NTHREADS)
void enc_main(const float* __restrict__ x, const float* __restrict__ codes,
              const float* __restrict__ scale, float* __restrict__ Eacc,
              float* __restrict__ sumW) {
    extern __shared__ char smem[];
    __bf16* codesb = (__bf16*)(smem + OFF_CODES);
    float*  xf     = (float*)(smem + OFF_XF);      // f32 staging (2 buffers)
    __bf16* xsb    = (__bf16*)(smem + OFF_XSB);    // x tile, row-major
    __bf16* xsbT   = (__bf16*)(smem + OFF_XSBT);   // x tile, d-major
    __bf16* WbT    = (__bf16*)(smem + OFF_WBT);    // softmax W, k-major
    float*  Wt     = (float*)(smem + OFF_WT);      // pre-softmax logits
    float*  c2     = (float*)(smem + OFF_C2);
    float*  sc     = (float*)(smem + OFF_SC);
    float*  x2     = (float*)(smem + OFF_X2);
    float*  swacc  = (float*)(smem + OFF_SW);

    const int tid  = threadIdx.x;
    const int wave = tid >> 5;
    const int lane = tid & 31;
    const int half = lane >> 4;
    const int l16  = lane & 15;

    const int b       = blockIdx.x >> 4;           // 16 chunks per batch
    const int rowbase = (blockIdx.x & 15) * CHUNK;
    const float* xbase = x + ((size_t)b * NN + rowbase) * DD;

#if HAVE_TDM
    // prologue: start DMA of row-block 0 while we convert the codebook
    if (tid < 32) tdm_load_tile(xbase, OFF_XF);
#endif

    // ---- preload codebook into LDS as bf16; precompute ||c_k||^2, scale ----
    for (int i = tid; i < (KK * DD) / 8; i += NTHREADS) {
        const float4 ca = ((const float4*)codes)[i * 2];
        const float4 cb = ((const float4*)codes)[i * 2 + 1];
        *(v8bf*)(codesb + i * 8) =
            v8bf{(__bf16)ca.x, (__bf16)ca.y, (__bf16)ca.z, (__bf16)ca.w,
                 (__bf16)cb.x, (__bf16)cb.y, (__bf16)cb.z, (__bf16)cb.w};
    }
    if (tid < KK) {
        float s = 0.0f;
        for (int d = 0; d < DD; ++d) {
            const float v = codes[tid * DD + d];
            s += v * v;
        }
        c2[tid]    = s;
        sc[tid]    = scale[tid];
        swacc[tid] = 0.0f;
    }
    __syncthreads();

    // ---- persistent GEMM2 accumulators: wave owns k-tile (wave&3) and
    //      d-columns [ (wave>>2)*256 , +256 ) as 16 tiles of 16x16 f32 ----
    const int kt2   = wave & 3;
    const int dbase = (wave >> 2) * 256;
    v8f acc2[16];
#pragma unroll
    for (int t = 0; t < 16; ++t) acc2[t] = v8f{};

    // softmax ownership: 8 threads per row, 8 k-values per thread
    const int srow = tid >> 3;
    const int kg0  = (tid & 7) * 8;
    float swreg[8];
#pragma unroll
    for (int j = 0; j < 8; ++j) swreg[j] = 0.0f;

    for (int it = 0; it < ITERS; ++it) {
        if (tid < ROWS) x2[tid] = 0.0f;

#if HAVE_TDM
        const int buf = it & 1;
        if (tid < 32) __builtin_amdgcn_s_wait_tensorcnt(0);  // tile `it` landed
        __syncthreads();                                     // visible to all waves
        if (it + 1 < ITERS && tid < 32)                      // prefetch next tile
            tdm_load_tile(xbase + (size_t)(it + 1) * ROWS * DD,
                          OFF_XF + (unsigned)(buf ^ 1) * (ROWS * DD * 4));
#else
        const int buf = 0;
        __syncthreads();
        // synchronous staging fallback: global -> f32 LDS tile
#pragma unroll
        for (int j = 0; j < 16; ++j) {
            const int idx4 = j * NTHREADS + tid;   // 4096 float4s
            ((float4*)xf)[idx4] =
                ((const float4*)(xbase + (size_t)it * ROWS * DD))[idx4];
        }
        if (it + 1 < ITERS)
            __builtin_prefetch(xbase + (size_t)(it + 1) * ROWS * DD + tid * 64, 0, 0);
        __syncthreads();
#endif
        const float* src = xf + (size_t)buf * (ROWS * DD);

        // ---- phase A: xsb (row-major bf16) + row sq-norms -----------------
        {
            const int row = tid >> 3;              // 64 elems/thread, one row
            const int ds0 = (tid & 7) * 64;
            float l2 = 0.0f;
#pragma unroll
            for (int j = 0; j < 64; j += 8) {
                const float4 va = *(const float4*)(src + row * DD + ds0 + j);
                const float4 vb = *(const float4*)(src + row * DD + ds0 + j + 4);
                l2 += va.x * va.x + va.y * va.y + va.z * va.z + va.w * va.w;
                l2 += vb.x * vb.x + vb.y * vb.y + vb.z * vb.z + vb.w * vb.w;
                *(v8bf*)(xsb + row * DD + ds0 + j) =
                    v8bf{(__bf16)va.x, (__bf16)va.y, (__bf16)va.z, (__bf16)va.w,
                         (__bf16)vb.x, (__bf16)vb.y, (__bf16)vb.z, (__bf16)vb.w};
            }
            atomicAdd(&x2[row], l2);
        }
        // ---- phase B: xsbT (d-major bf16) via 8-row gather, 16B stores ----
#pragma unroll
        for (int j = 0; j < 8; ++j) {
            const int idx = j * NTHREADS + tid;    // 2048 (d, octet) pairs
            const int d   = idx & 511;
            const int oct = idx >> 9;              // 0..3 (rows oct*8..+8)
            const float* col = src + (size_t)oct * 8 * DD + d;
            v8bf t;
#pragma unroll
            for (int r = 0; r < 8; ++r) t[r] = (__bf16)col[(size_t)r * DD];
            *(v8bf*)(xsbT + d * ROWS + oct * 8) = t;
        }
        __syncthreads();

        // ---- GEMM1 (pipelined): one 16x16 tile of x . codes^T per wave ----
        {
            const int mt   = wave & 1;
            const int kt   = wave >> 1;
            const int arow = (mt * 16 + l16) * DD;
            const int kcol = (kt * 16 + l16) * DD;
            v8f acc = v8f{};
            v16bf a0 = ldfrag(xsb, arow, half);
            v16bf b0 = ldfrag(codesb, kcol, half);
#pragma unroll
            for (int dsr = 32; dsr < DD; dsr += 32) {
                const v16bf a1 = ldfrag(xsb, arow + dsr, half);
                const v16bf b1 = ldfrag(codesb, kcol + dsr, half);
                acc = __builtin_amdgcn_wmma_f32_16x16x32_bf16(
                    false, a0, false, b0, (short)0, acc, false, false);
                a0 = a1; b0 = b1;
            }
            acc = __builtin_amdgcn_wmma_f32_16x16x32_bf16(
                false, a0, false, b0, (short)0, acc, false, false);
            // dist = sqrt(max(x2 + c2 - 2 dot, eps)); logit = dist * scale[k]
#pragma unroll
            for (int r = 0; r < 8; ++r) {
                const int row = mt * 16 + r + 8 * half;
                const int k   = kt * 16 + l16;
                float sq = x2[row] + c2[k] - 2.0f * acc[r];
                sq = fmaxf(sq, EPS);
                Wt[row * KK + k] = sqrtf(sq) * sc[k];
            }
        }
        __syncthreads();

        // ---- softmax over K=64: 8 threads/row, shuffle reductions ---------
        {
            const float4 ta = *(const float4*)(Wt + srow * KK + kg0);
            const float4 tb = *(const float4*)(Wt + srow * KK + kg0 + 4);
            float e[8] = {ta.x, ta.y, ta.z, ta.w, tb.x, tb.y, tb.z, tb.w};
            float mx = e[0];
#pragma unroll
            for (int j = 1; j < 8; ++j) mx = fmaxf(mx, e[j]);
            mx = fmaxf(mx, __shfl_xor(mx, 1));
            mx = fmaxf(mx, __shfl_xor(mx, 2));
            mx = fmaxf(mx, __shfl_xor(mx, 4));
            float s = 0.0f;
#pragma unroll
            for (int j = 0; j < 8; ++j) { e[j] = __expf(e[j] - mx); s += e[j]; }
            s += __shfl_xor(s, 1);
            s += __shfl_xor(s, 2);
            s += __shfl_xor(s, 4);
            const float inv = 1.0f / s;
#pragma unroll
            for (int j = 0; j < 8; ++j) {
                const float wv = e[j] * inv;
                WbT[(kg0 + j) * ROWS + srow] = (__bf16)wv;
                swreg[j] += wv;            // per-thread running sum_i W[i,k]
            }
        }
        __syncthreads();

        // ---- GEMM2 (pipelined): acc2 += W^T (16k x 32i) x xs (32i x 16d) --
        {
            const int krow = kt2 * 16 + l16;
            const v16bf a2 = ldfrag(WbT, krow * ROWS, half);
            v16bf bx0 = ldfrag(xsbT, (dbase + l16) * ROWS, half);
#pragma unroll
            for (int dt = 1; dt < 16; ++dt) {
                const v16bf bx1 = ldfrag(xsbT, (dbase + dt * 16 + l16) * ROWS, half);
                acc2[dt - 1] = __builtin_amdgcn_wmma_f32_16x16x32_bf16(
                    false, a2, false, bx0, (short)0, acc2[dt - 1], false, false);
                bx0 = bx1;
            }
            acc2[15] = __builtin_amdgcn_wmma_f32_16x16x32_bf16(
                false, a2, false, bx0, (short)0, acc2[15], false, false);
        }
        __syncthreads();   // LDS tiles reused next iteration
    }

    // ---- cross-chunk reduction: atomic f32 adds into global E / sumW ----
#pragma unroll
    for (int j = 0; j < 8; ++j) atomicAdd(&swacc[kg0 + j], swreg[j]);
#pragma unroll
    for (int dt = 0; dt < 16; ++dt) {
        const int dcol = dbase + dt * 16 + l16;
#pragma unroll
        for (int r = 0; r < 8; ++r) {
            const int k = kt2 * 16 + r + 8 * half;
            atomicAdd(&Eacc[((size_t)b * KK + k) * DD + dcol], acc2[dt][r]);
        }
    }
    __syncthreads();
    if (tid < KK) atomicAdd(&sumW[b * KK + tid], swacc[tid]);
}

// --------------------------------------------------------------------------
// Kernel 2: E = (E - sumW * codes), then L2-normalize over D (in-place).
// --------------------------------------------------------------------------
extern "C" __global__ __launch_bounds__(256)
void enc_finalize(float* __restrict__ E, const float* __restrict__ codes,
                  const float* __restrict__ sumW) {
    __shared__ float red[256];
    __shared__ float rn_s;
    const int bk = blockIdx.x;          // b*64 + k
    const int k  = bk & 63;
    const float sw = sumW[bk];
    float* base = E + (size_t)bk * DD;
    const int d0 = threadIdx.x, d1 = threadIdx.x + 256;
    const float v0 = base[d0] - sw * codes[k * DD + d0];
    const float v1 = base[d1] - sw * codes[k * DD + d1];
    red[threadIdx.x] = v0 * v0 + v1 * v1;
    __syncthreads();
#pragma unroll
    for (int s = 128; s > 0; s >>= 1) {
        if (threadIdx.x < s) red[threadIdx.x] += red[threadIdx.x + s];
        __syncthreads();
    }
    if (threadIdx.x == 0) rn_s = rsqrtf(fmaxf(red[0], EPS));
    __syncthreads();
    const float rn = rn_s;
    base[d0] = v0 * rn;
    base[d1] = v1 * rn;
}

// --------------------------------------------------------------------------
extern "C" void kernel_launch(void* const* d_in, const int* in_sizes, int n_in,
                              void* d_out, int out_size, void* d_ws, size_t ws_size,
                              hipStream_t stream) {
    const float* x     = (const float*)d_in[0];
    const float* codes = (const float*)d_in[1];
    const float* scale = (const float*)d_in[2];
    float* E    = (float*)d_out;          // (B,K,D) f32 accumulator + result
    float* sumW = (float*)d_ws;           // (B,K) f32

    (void)hipFuncSetAttribute((const void*)enc_main,
                              hipFuncAttributeMaxDynamicSharedMemorySize,
                              SHMEM_BYTES);

    enc_zero<<<256, 256, 0, stream>>>(E, sumW, BB * KK * DD, BB * KK);
    enc_main<<<BB * (NN / CHUNK), NTHREADS, SHMEM_BYTES, stream>>>(x, codes, scale, E, sumW);
    enc_finalize<<<BB * KK, 256, 0, stream>>>(E, codes, sumW);
}
